// CPCModel_NCE_45930380263763
// MI455X (gfx1250) — matrix operand order repaired and checked by reading, not compiled, not run on hardware
//
#include <hip/hip_runtime.h>
#include <hip/hip_bf16.h>
#include <cstdint>

// ---------------------------------------------------------------------------
// CPC/NCE pipeline for MI455X (gfx1250, wave32, WMMA, TDM).
//   conv1 (Cin=1)            : direct VALU kernel (tiny FLOPs, store-bound)
//   conv2..5                 : implicit-im2col GEMM on v_wmma_f32_16x16x32_f16
//   BN (post-ReLU, over B,T) : block-per-channel LDS reduction + apply
//   GRU input proj           : WMMA GEMM, TDM double-buffered LDS staging
//   GRU recurrence           : persistent 32-wave workgroup, h + gh in LDS
//   coupling est (12x)       : WMMA GEMM, TDM double-buffered LDS staging
//   scores / logsoftmax / labels : VALU kernels writing straight into d_out
// Precision: f16 WMMA operands, f32 accumulate.
// B operands are kept N-major (their native layout: wih 768x512, whh 768x256,
// coup_w[k] 512x256) so every B fragment is 16 contiguous f16 per lane
// (-> b128 loads), matching how A fragments already coalesce.
// ---------------------------------------------------------------------------

typedef __attribute__((ext_vector_type(16))) _Float16 v16h;
typedef __attribute__((ext_vector_type(8)))  float    v8f;
typedef unsigned int v4u __attribute__((ext_vector_type(4)));
typedef int          v8i __attribute__((ext_vector_type(8)));
typedef int          v4i __attribute__((ext_vector_type(4)));

#define BN_EPS 1e-5f

#if __has_builtin(__builtin_amdgcn_tensor_load_to_lds) && \
    __has_builtin(__builtin_amdgcn_s_wait_tensorcnt)
#define HAVE_TDM 1
#else
#define HAVE_TDM 0
#endif

// ---- WMMA fragment helpers (A/B in contiguous k order; D per ISA 7.12.2) ---------------------------

// A fragment: 16x32 f16 tile of row-major f16 matrix.
// lanes 0-15  hold row M=lane,    K = {0..15}
// lanes 16-31 hold row M=lane-16, K = {16..31}
__device__ __forceinline__ v16h load_a_frag_f16(const _Float16* A, int lda,
                                                int row0, int k0) {
  const int lane = threadIdx.x & 31;
  const int r  = lane & 15;
  const int hi = lane >> 4;
  const _Float16* p = A + (size_t)(row0 + r) * lda + k0;
  const int kb0 = hi ? 16 : 0;
  const int kb1 = kb0 + 8;
  v16h a;
#pragma unroll
  for (int i = 0; i < 8; ++i) a[i] = p[kb0 + i];
#pragma unroll
  for (int i = 0; i < 8; ++i) a[8 + i] = p[kb1 + i];
  return a;
}

// Same A fragment but sourced from an f32 LDS tile (convert on load).
__device__ __forceinline__ v16h load_a_frag_lds_f32(const float* H, int ldh,
                                                    int row0, int k0) {
  const int lane = threadIdx.x & 31;
  const int r  = lane & 15;
  const int hi = lane >> 4;
  const float* p = H + (row0 + r) * ldh + k0;
  const int kb0 = hi ? 16 : 0;
  const int kb1 = kb0 + 8;
  v16h a;
#pragma unroll
  for (int i = 0; i < 8; ++i) a[i] = (_Float16)p[kb0 + i];
#pragma unroll
  for (int i = 0; i < 8; ++i) a[8 + i] = (_Float16)p[kb1 + i];
  return a;
}

// B fragment from an N-major (N x K row-major, i.e. B-transposed) source.
// VGPR v: lanes 0-15 hold (K=2v,2v+1, N=lane); lanes 16-31 K+=16.
// Per lane this is 16 contiguous f16 -> two b128 loads.
__device__ __forceinline__ v16h load_bT_frag_f16(const _Float16* BT, int ldk,
                                                 int k0, int col0) {
  const int lane = threadIdx.x & 31;
  const int c    = lane & 15;
  const int koff = (lane >> 4) << 4;  // 0 or 16
  const _Float16* p = BT + (size_t)(col0 + c) * ldk + k0 + koff;
  v16h b;
#pragma unroll
  for (int i = 0; i < 16; ++i) b[i] = p[i];
  return b;
}

// D/C 16x16 f32: VGPR j: lanes 0-15 -> (M=j, N=lane); lanes 16-31 -> (M=j+8).
__device__ __forceinline__ void store_d_tile(float* out, int ldo,
                                             int row0, int col0, v8f acc) {
  const int lane = threadIdx.x & 31;
  const int n  = col0 + (lane & 15);
  const int mb = row0 + ((lane >> 4) << 3);
#pragma unroll
  for (int j = 0; j < 8; ++j) out[(size_t)(mb + j) * ldo + n] = acc[j];
}

#if HAVE_TDM
// ---- Tensor Data Mover: 2D f16 tile (global, row-major) -> LDS -----------
// D# per ISA 8.3-8.4. data_size=1 (2 bytes). Tile rows are packed
// contiguously in LDS (tile_dim0 elements per row). Extents are given
// relative to the tile origin so no OOB clipping occurs.
__device__ __forceinline__ void tdm_load_2d_f16(unsigned lds_off,
                                                const _Float16* gptr,
                                                unsigned tile_w, unsigned tile_h,
                                                unsigned w_rem, unsigned h_rem,
                                                unsigned long long row_stride) {
  // group0: count=1 | lds_addr<<32 ; global_addr(57b) | type=2<<62
  unsigned long long q0 = 1ull | ((unsigned long long)lds_off << 32);
  unsigned long long ga = (unsigned long long)(uintptr_t)gptr;
  unsigned long long q1 = (ga & ((1ull << 57) - 1)) | (2ull << 62);
  v4u g0;
  g0[0] = (unsigned)q0; g0[1] = (unsigned)(q0 >> 32);
  g0[2] = (unsigned)q1; g0[3] = (unsigned)(q1 >> 32);
  // group1: [17:16]=data_size(2B) [79:48]=tensor_dim0 [111:80]=tensor_dim1
  //         [127:112]=tile_dim0 [143:128]=tile_dim1 [207:160]=dim0_stride
  unsigned long long p0 = (1ull << 16) |
                          ((unsigned long long)(w_rem & 0xFFFFu) << 48);
  unsigned long long p1 = (unsigned long long)(w_rem >> 16) |
                          ((unsigned long long)h_rem << 16) |
                          ((unsigned long long)(tile_w & 0xFFFFu) << 48);
  unsigned long long p2 = (unsigned long long)(tile_h & 0xFFFFu) |
                          ((row_stride & 0xFFFFFFFFull) << 32);
  unsigned long long p3 = (row_stride >> 32) & 0xFFFFull;
  v8i g1;
  g1[0] = (int)p0; g1[1] = (int)(p0 >> 32);
  g1[2] = (int)p1; g1[3] = (int)(p1 >> 32);
  g1[4] = (int)p2; g1[5] = (int)(p2 >> 32);
  g1[6] = (int)p3; g1[7] = (int)(p3 >> 32);
  v4i gz4 = {0, 0, 0, 0};
#if defined(__clang_major__) && (__clang_major__ >= 23)
  v8i gz8 = {0, 0, 0, 0, 0, 0, 0, 0};
  __builtin_amdgcn_tensor_load_to_lds(g0, g1, gz4, gz4, gz8, 0);
#else
  __builtin_amdgcn_tensor_load_to_lds(g0, g1, gz4, gz4, 0);
#endif
}

__device__ __forceinline__ unsigned lds_off_of(const void* p) {
  // flat shared address: LDS byte offset lives in addr[31:0]
  return (unsigned)(size_t)p;
}
#endif  // HAVE_TDM

// ---- conv1 (Cin=1, Kw=10, S=5, P=2): direct ------------------------------

__global__ void conv1_direct(const float* __restrict__ x,
                             const float* __restrict__ w,
                             const float* __restrict__ bias,
                             float* __restrict__ y, int Lout) {
  int idx = blockIdx.x * blockDim.x + threadIdx.x;
  int total = 64 * 512 * Lout;
  if (idx >= total) return;
  int t  = idx % Lout;
  int co = (idx / Lout) % 512;
  int b  = idx / (Lout * 512);
  const float* xb = x + (size_t)b * 20480;
  const float* wc = w + co * 10;
  float acc = bias[co];
  int p0 = t * 5 - 2;
#pragma unroll
  for (int kw = 0; kw < 10; ++kw) {
    int p = p0 + kw;
    if (p >= 0 && p < 20480) acc += wc[kw] * xb[p];
  }
  y[idx] = acc > 0.f ? acc : 0.f;  // ReLU fused
}

// ---- conv2..5 as implicit-im2col WMMA GEMM -------------------------------
// M = 512 (Cout), N = 64*Lout (cols = b*Lout+t), K = Ci*Kw.
// Block: 256 threads = 8 waves as 4(M) x 2(N) -> 64x32 block tile.

template <int KW_LOG2>
__global__ void conv_gemm_wmma(const float* __restrict__ X,
                               const _Float16* __restrict__ W16,
                               const float* __restrict__ bias,
                               float* __restrict__ Y,
                               int Ci, int Lin, int Lout, int S, int P) {
  const int wave = threadIdx.x >> 5;
  const int lane = threadIdx.x & 31;
  const int mt = wave & 3;
  const int nt = wave >> 2;
  const int co0  = blockIdx.y * 64 + mt * 16;
  const int col0 = blockIdx.x * 32 + nt * 16;
  const int K   = Ci << KW_LOG2;
  const int KWm = (1 << KW_LOG2) - 1;

  const int col = col0 + (lane & 15);
  const int bb  = col / Lout;
  const int tt  = col - bb * Lout;
  const int pbase = tt * S - P;
  const float* Xb = X + (size_t)bb * Ci * Lin;

  v8f acc = {};
  {
    const int mr = (lane >> 4) << 3;
#pragma unroll
    for (int j = 0; j < 8; ++j) acc[j] = bias[co0 + mr + j];
  }
  const int koff = (lane >> 4) << 4;
  for (int kc = 0; kc < K; kc += 32) {
    if (kc + 32 < K)  // global_prefetch_b8 on next weight panel
      __builtin_prefetch(W16 + (size_t)(co0 + (lane & 15)) * K + kc + 32, 0, 3);
    v16h a = load_a_frag_f16(W16, K, co0, kc);
    v16h b;
#pragma unroll
    for (int v = 0; v < 8; ++v) {
#pragma unroll
      for (int u = 0; u < 2; ++u) {
        int k  = kc + koff + 2 * v + u;
        int ci = k >> KW_LOG2;
        int kw = k & KWm;
        int pos = pbase + kw;
        float val = (pos >= 0 && pos < Lin) ? Xb[(size_t)ci * Lin + pos] : 0.f;
        b[2 * v + u] = (_Float16)val;
      }
    }
    acc = __builtin_amdgcn_wmma_f32_16x16x32_f16(false, a, false, b,
                                                 (short)0, acc, false, false);
  }
  const int mb = co0 + ((lane >> 4) << 3);
#pragma unroll
  for (int j = 0; j < 8; ++j) {
    float r = acc[j] > 0.f ? acc[j] : 0.f;
    Y[((size_t)bb * 512 + (mb + j)) * Lout + tt] = r;
  }
}

// ---- BatchNorm over (B, T) per channel (post-ReLU, biased var) -----------

__global__ void bn_stats(const float* __restrict__ act, float* __restrict__ stats,
                         int Lout) {
  __shared__ float ssum[256];
  __shared__ float ssq[256];
  const int c = blockIdx.x;
  const int n = 64 * Lout;
  float s = 0.f, q = 0.f;
  for (int i = threadIdx.x; i < n; i += 256) {
    int b = i / Lout, t = i - b * Lout;
    float v = act[((size_t)b * 512 + c) * Lout + t];
    s += v; q += v * v;
  }
  ssum[threadIdx.x] = s; ssq[threadIdx.x] = q;
  __syncthreads();
  for (int off = 128; off > 0; off >>= 1) {
    if (threadIdx.x < off) {
      ssum[threadIdx.x] += ssum[threadIdx.x + off];
      ssq[threadIdx.x]  += ssq[threadIdx.x + off];
    }
    __syncthreads();
  }
  if (threadIdx.x == 0) {
    float mu = ssum[0] / n;
    stats[c]       = mu;
    stats[512 + c] = ssq[0] / n - mu * mu;
  }
}

__global__ void bn_apply(float* __restrict__ act, const float* __restrict__ stats,
                         const float* __restrict__ g, const float* __restrict__ beta,
                         int Lout) {
  size_t idx = (size_t)blockIdx.x * blockDim.x + threadIdx.x;
  size_t total = (size_t)64 * 512 * Lout;
  if (idx >= total) return;
  int c = (int)((idx / Lout) % 512);
  float v = act[idx];
  act[idx] = g[c] * (v - stats[c]) * rsqrtf(stats[512 + c] + BN_EPS) + beta[c];
}

// ---- layout/precision shuffles -------------------------------------------

__global__ void feats_to_z(const float* __restrict__ feats,
                           float* __restrict__ z32, _Float16* __restrict__ z16) {
  int idx = blockIdx.x * blockDim.x + threadIdx.x;  // 64*128*512
  if (idx >= 64 * 128 * 512) return;
  int c = idx % 512;
  int t = (idx / 512) % 128;
  int b = idx / (512 * 128);
  float v = feats[((size_t)b * 512 + c) * 128 + t];
  z32[idx] = v;
  z16[idx] = (_Float16)v;
}

__global__ void convert_f32_to_f16(const float* __restrict__ src,
                                   _Float16* __restrict__ dst, int n) {
  int i = blockIdx.x * blockDim.x + threadIdx.x;
  if (i < n) dst[i] = (_Float16)src[i];
}

// ---- generic WMMA GEMM: D = A(MxK) @ BT(NxK)^T + bias[N] -----------------
// TDM path: wave 0 drives two tensor_load_to_lds per 32-K chunk (A panel
// 64x32, BT panel 32x32, f16), double buffered; s_wait_tensorcnt + workgroup
// barrier hand the panels to the 8 WMMA waves, which read them via ds_load
// (contiguous 16-f16 per lane on both operands -> b128 LDS reads).

__global__ void gemm_wmma_bT(const _Float16* __restrict__ A,
                             const _Float16* __restrict__ BT,
                             const float* __restrict__ bias,
                             float* __restrict__ D,
                             int M, int N, int K) {
  const int wave = threadIdx.x >> 5;
  const int lane = threadIdx.x & 31;
  const int mt = wave & 3, nt = wave >> 2;
  const int rowB = blockIdx.y * 64;
  const int colB = blockIdx.x * 32;
  const int row0 = rowB + mt * 16;
  const int col0 = colB + nt * 16;
  v8f acc = {};
  {
    float bv = bias ? bias[col0 + (lane & 15)] : 0.f;
#pragma unroll
    for (int j = 0; j < 8; ++j) acc[j] = bv;
  }
#if HAVE_TDM
  __shared__ _Float16 sA[2][64 * 32];
  __shared__ _Float16 sBT[2][32 * 32];
  if (wave == 0) {
    tdm_load_2d_f16(lds_off_of(&sA[0][0]), A + (size_t)rowB * K,
                    32, 64, (unsigned)K, (unsigned)(M - rowB),
                    (unsigned long long)K);
    tdm_load_2d_f16(lds_off_of(&sBT[0][0]), BT + (size_t)colB * K,
                    32, 32, (unsigned)K, (unsigned)(N - colB),
                    (unsigned long long)K);
    __builtin_amdgcn_s_wait_tensorcnt(0);
  }
  __syncthreads();
  int cur = 0;
  for (int kc = 0; kc < K; kc += 32) {
    const int nxt = cur ^ 1;
    if ((kc + 32 < K) && wave == 0) {  // prefetch next panels via TDM
      tdm_load_2d_f16(lds_off_of(&sA[nxt][0]),
                      A + (size_t)rowB * K + (kc + 32),
                      32, 64, (unsigned)(K - kc - 32), (unsigned)(M - rowB),
                      (unsigned long long)K);
      tdm_load_2d_f16(lds_off_of(&sBT[nxt][0]),
                      BT + (size_t)colB * K + (kc + 32),
                      32, 32, (unsigned)(K - kc - 32), (unsigned)(N - colB),
                      (unsigned long long)K);
    }
    v16h a = load_a_frag_f16(sA[cur], 32, mt * 16, 0);
    v16h b = load_bT_frag_f16(sBT[cur], 32, 0, nt * 16);
    acc = __builtin_amdgcn_wmma_f32_16x16x32_f16(false, a, false, b,
                                                 (short)0, acc, false, false);
    if (wave == 0) __builtin_amdgcn_s_wait_tensorcnt(0);
    __syncthreads();
    cur = nxt;
  }
#else
  (void)M;
  for (int kc = 0; kc < K; kc += 32) {
    if (kc + 32 < K) {
      __builtin_prefetch(A + (size_t)(row0 + (lane & 15)) * K + kc + 32, 0, 3);
      __builtin_prefetch(BT + (size_t)(col0 + (lane & 15)) * K + kc + 32, 0, 3);
    }
    v16h a = load_a_frag_f16(A, K, row0, kc);
    v16h b = load_bT_frag_f16(BT, K, kc, col0);
    acc = __builtin_amdgcn_wmma_f32_16x16x32_f16(false, a, false, b,
                                                 (short)0, acc, false, false);
  }
#endif
  store_d_tile(D, N, row0, col0, acc);
}

// ---- GRU recurrence: persistent single-WGP kernel ------------------------
// LDS: h (64x256 f32, 64KB) + gh (64x768 f32, 192KB) = 256KB of 320KB/WGP.
// whh16 is N-major (768x256): contiguous 16-f16 B fragments from global.

__global__ void __launch_bounds__(1024)
gru_wmma(const float* __restrict__ gates_i, const _Float16* __restrict__ whh16,
         const float* __restrict__ bhh, _Float16* __restrict__ ct16) {
  extern __shared__ float smem[];
  float* h  = smem;             // 64 x 256
  float* gh = smem + 64 * 256;  // 64 x 768
  const int tid  = threadIdx.x;
  const int wave = tid >> 5;
  const int lane = tid & 31;
  for (int i = tid; i < 64 * 256; i += 1024) h[i] = 0.f;
  __syncthreads();
  for (int t = 0; t < 128; ++t) {
    for (int tile = wave; tile < 192; tile += 32) {  // 4 x 48 tiles of 16x16
      const int mt = tile / 48;
      const int nt = tile % 48;
      const int row0 = mt * 16, col0 = nt * 16;
      v8f acc;
      {
        float bv = bhh[col0 + (lane & 15)];
#pragma unroll
        for (int j = 0; j < 8; ++j) acc[j] = bv;
      }
#pragma unroll
      for (int kc = 0; kc < 256; kc += 32) {
        if (kc + 32 < 256)
          __builtin_prefetch(whh16 + (size_t)(col0 + (lane & 15)) * 256 + kc + 32,
                             0, 3);
        v16h a = load_a_frag_lds_f32(h, 256, row0, kc);
        v16h b = load_bT_frag_f16(whh16, 256, kc, col0);
        acc = __builtin_amdgcn_wmma_f32_16x16x32_f16(false, a, false, b,
                                                     (short)0, acc, false, false);
      }
      const int n  = col0 + (lane & 15);
      const int mb = row0 + ((lane >> 4) << 3);
#pragma unroll
      for (int j = 0; j < 8; ++j) gh[(mb + j) * 768 + n] = acc[j];
    }
    __syncthreads();
    for (int i = tid; i < 64 * 256; i += 1024) {
      const int b = i >> 8, hh = i & 255;
      const float* gi  = gates_i + ((size_t)b * 128 + t) * 768;
      const float* ghb = gh + b * 768;
      float r  = 1.f / (1.f + expf(-(gi[hh]       + ghb[hh])));
      float u  = 1.f / (1.f + expf(-(gi[256 + hh] + ghb[256 + hh])));
      float nn = tanhf(gi[512 + hh] + r * ghb[512 + hh]);
      float hv = (1.f - u) * nn + u * h[i];
      h[i] = hv;
      ct16[((size_t)b * 128 + t) * 256 + hh] = (_Float16)hv;
    }
    __syncthreads();
  }
}

// ---- NCE scores / loss / labels ------------------------------------------
// score(b,s,tt) = est(b,k+tt) . (s==0 ? z[b,k+tt] : z[(b-s)%64, perm[k+tt]])

__global__ void scores_kernel(const float* __restrict__ est,
                              const float* __restrict__ z,
                              const int* __restrict__ perm,
                              float* __restrict__ logits, int k, int lk) {
  int idx = blockIdx.x * blockDim.x + threadIdx.x;
  int total = 64 * 9 * lk;
  if (idx >= total) return;
  int tt = idx % lk;
  int s  = (idx / lk) % 9;
  int b  = idx / (9 * lk);
  const float* e = est + ((size_t)b * 128 + (k + tt)) * 512;
  int zb, zt;
  if (s == 0) { zb = b; zt = k + tt; }
  else        { zb = (b - s) & 63; zt = perm[k + tt]; }
  const float* zr = z + ((size_t)zb * 128 + zt) * 512;
  float acc = 0.f;
#pragma unroll 8
  for (int c = 0; c < 512; ++c) acc += e[c] * zr[c];
  logits[idx] = acc;
}

// Reference softmaxes consecutive groups of 9 of the flattened (b,9,l) array.
__global__ void loss_kernel(const float* __restrict__ logits,
                            float* __restrict__ loss, int k, int lk) {
  int row = blockIdx.x * blockDim.x + threadIdx.x;
  if (row >= 64 * lk) return;
  const float* p = logits + (size_t)row * 9;
  float mx = p[0];
#pragma unroll
  for (int j = 1; j < 9; ++j) mx = fmaxf(mx, p[j]);
  float se = 0.f, sx = 0.f;
#pragma unroll
  for (int j = 0; j < 9; ++j) { se += expf(p[j] - mx); sx += p[j]; }
  float lse = mx + logf(se);
  atomicAdd(loss, (9.f * lse - sx) / ((float)(128 - k) * 64.f));
}

__global__ void labels_kernel(float* __restrict__ labels, int lk) {
  int idx = blockIdx.x * blockDim.x + threadIdx.x;
  if (idx >= 64 * 9 * lk) return;
  int s = (idx / lk) % 9;
  labels[idx] = (s == 0) ? 1.f : 0.f;
}

__global__ void zero_kernel(float* __restrict__ p, int n) {
  int i = blockIdx.x * blockDim.x + threadIdx.x;
  if (i < n) p[i] = 0.f;
}

// ---------------------------------------------------------------------------

extern "C" void kernel_launch(void* const* d_in, const int* in_sizes, int n_in,
                              void* d_out, int out_size, void* d_ws, size_t ws_size,
                              hipStream_t stream) {
  (void)in_sizes; (void)n_in; (void)out_size; (void)ws_size;
  const float* x    = (const float*)d_in[0];
  const int*   perm = (const int*)d_in[1];
  const float* convw[5]; const float* convb[5]; const float* bng[5]; const float* bnb[5];
  for (int i = 0; i < 5; ++i) {
    convw[i] = (const float*)d_in[2 + i];
    convb[i] = (const float*)d_in[7 + i];
    bng[i]   = (const float*)d_in[12 + i];
    bnb[i]   = (const float*)d_in[17 + i];
  }
  const float* wih   = (const float*)d_in[22];
  const float* whh   = (const float*)d_in[23];
  const float* bih   = (const float*)d_in[24];
  const float* bhh   = (const float*)d_in[25];
  const float* coupw = (const float*)d_in[26];
  const float* coupb = (const float*)d_in[27];
  float* out = (float*)d_out;

  char* ws = (char*)d_ws;
  size_t off = 0;
  auto alloc = [&](size_t bytes) -> void* {
    void* p = (void*)(ws + off);
    off += (bytes + 255) & ~(size_t)255;
    return p;
  };
  float*    actA   = (float*)alloc((size_t)64 * 512 * 4095 * 4);
  float*    actB   = (float*)alloc((size_t)64 * 512 * 1023 * 4);
  float*    feats  = (float*)alloc((size_t)64 * 512 * 128 * 4);
  float*    z32    = (float*)alloc((size_t)64 * 128 * 512 * 4);
  _Float16* z16    = (_Float16*)alloc((size_t)64 * 128 * 512 * 2);
  float*    gatesI = (float*)alloc((size_t)64 * 128 * 768 * 4);
  _Float16* ct16   = (_Float16*)alloc((size_t)64 * 128 * 256 * 2);
  float*    est    = (float*)alloc((size_t)8192 * 512 * 4);
  _Float16* w16_2  = (_Float16*)alloc((size_t)512 * 4096 * 2);
  _Float16* w16_3  = (_Float16*)alloc((size_t)512 * 2048 * 2);
  _Float16* w16_4  = (_Float16*)alloc((size_t)512 * 2048 * 2);
  _Float16* w16_5  = (_Float16*)alloc((size_t)512 * 2048 * 2);
  _Float16* wih16  = (_Float16*)alloc((size_t)768 * 512 * 2);
  _Float16* whh16  = (_Float16*)alloc((size_t)768 * 256 * 2);
  _Float16* coup16 = (_Float16*)alloc((size_t)12 * 512 * 256 * 2);
  float*    stats  = (float*)alloc(1024 * 4);

  zero_kernel<<<1, 32, 0, stream>>>(out, 12);

  // weight packing: f32 -> f16, keep native (N-major) layouts
  convert_f32_to_f16<<<(512 * 4096 + 255) / 256, 256, 0, stream>>>(convw[1], w16_2, 512 * 4096);
  convert_f32_to_f16<<<(512 * 2048 + 255) / 256, 256, 0, stream>>>(convw[2], w16_3, 512 * 2048);
  convert_f32_to_f16<<<(512 * 2048 + 255) / 256, 256, 0, stream>>>(convw[3], w16_4, 512 * 2048);
  convert_f32_to_f16<<<(512 * 2048 + 255) / 256, 256, 0, stream>>>(convw[4], w16_5, 512 * 2048);
  convert_f32_to_f16<<<(768 * 512 + 255) / 256, 256, 0, stream>>>(wih, wih16, 768 * 512);
  convert_f32_to_f16<<<(768 * 256 + 255) / 256, 256, 0, stream>>>(whh, whh16, 768 * 256);
  convert_f32_to_f16<<<(12 * 512 * 256 + 255) / 256, 256, 0, stream>>>(coupw, coup16, 12 * 512 * 256);

  {  // layer 1: 20480 -> 4095
    int Lout = 4095, total = 64 * 512 * Lout;
    conv1_direct<<<(total + 255) / 256, 256, 0, stream>>>(x, convw[0], convb[0], actA, Lout);
    bn_stats<<<512, 256, 0, stream>>>(actA, stats, Lout);
    bn_apply<<<(total + 255) / 256, 256, 0, stream>>>(actA, stats, bng[0], bnb[0], Lout);
  }
  {  // layer 2: 4095 -> 1023 (Kw=8,S=4,P=2)
    int Lout = 1023, total = 64 * 512 * Lout;
    conv_gemm_wmma<3><<<dim3(64 * Lout / 32, 8), 256, 0, stream>>>(
        actA, w16_2, convb[1], actB, 512, 4095, Lout, 4, 2);
    bn_stats<<<512, 256, 0, stream>>>(actB, stats, Lout);
    bn_apply<<<(total + 255) / 256, 256, 0, stream>>>(actB, stats, bng[1], bnb[1], Lout);
  }
  {  // layer 3: 1023 -> 512 (Kw=4,S=2,P=2)
    int Lout = 512, total = 64 * 512 * Lout;
    conv_gemm_wmma<2><<<dim3(64 * Lout / 32, 8), 256, 0, stream>>>(
        actB, w16_3, convb[2], actA, 512, 1023, Lout, 2, 2);
    bn_stats<<<512, 256, 0, stream>>>(actA, stats, Lout);
    bn_apply<<<(total + 255) / 256, 256, 0, stream>>>(actA, stats, bng[2], bnb[2], Lout);
  }
  {  // layer 4: 512 -> 257
    int Lout = 257, total = 64 * 512 * Lout;
    conv_gemm_wmma<2><<<dim3(64 * Lout / 32, 8), 256, 0, stream>>>(
        actA, w16_4, convb[3], actB, 512, 512, Lout, 2, 2);
    bn_stats<<<512, 256, 0, stream>>>(actB, stats, Lout);
    bn_apply<<<(total + 255) / 256, 256, 0, stream>>>(actB, stats, bng[3], bnb[3], Lout);
  }
  {  // layer 5: 257 -> 128 (P=1)
    int Lout = 128, total = 64 * 512 * Lout;
    conv_gemm_wmma<2><<<dim3(64 * Lout / 32, 8), 256, 0, stream>>>(
        actB, w16_5, convb[4], feats, 512, 257, Lout, 2, 1);
    bn_stats<<<512, 256, 0, stream>>>(feats, stats, Lout);
    bn_apply<<<(total + 255) / 256, 256, 0, stream>>>(feats, stats, bng[4], bnb[4], Lout);
  }

  feats_to_z<<<(64 * 128 * 512 + 255) / 256, 256, 0, stream>>>(feats, z32, z16);
  gemm_wmma_bT<<<dim3(768 / 32, 8192 / 64), 256, 0, stream>>>(
      z16, wih16, bih, gatesI, 8192, 768, 512);
  gru_wmma<<<1, 1024, (64 * 256 + 64 * 768) * sizeof(float), stream>>>(
      gatesI, whh16, bhh, ct16);

  const size_t logits_base = 12;
  const size_t labels_base = 12 + 839808;  // sum_k 64*9*(128-k)
  size_t lo = 0;
  for (int k = 1; k <= 12; ++k) {
    int lk = 128 - k;
    gemm_wmma_bT<<<dim3(512 / 32, 8192 / 64), 256, 0, stream>>>(
        ct16, coup16 + (size_t)(k - 1) * 512 * 256, coupb + (k - 1) * 512,
        est, 8192, 512, 256);
    int tot = 64 * 9 * lk;
    scores_kernel<<<(tot + 255) / 256, 256, 0, stream>>>(
        est, z32, perm, out + logits_base + lo, k, lk);
    loss_kernel<<<(64 * lk + 255) / 256, 256, 0, stream>>>(
        out + logits_base + lo, out + (k - 1), k, lk);
    labels_kernel<<<(tot + 255) / 256, 256, 0, stream>>>(
        out + labels_base + lo, lk);
    lo += (size_t)tot;
  }
}
